// BinaryBlock_25305947308063
// MI455X (gfx1250) — matrix operand order repaired and checked by reading, not compiled, not run on hardware
//
#include <hip/hip_runtime.h>
#include <hip/hip_bf16.h>

typedef __attribute__((ext_vector_type(16))) _Float16 v16h;
typedef __attribute__((ext_vector_type(8)))  _Float16 v8h;
typedef __attribute__((ext_vector_type(8)))  float    v8f;

#define B_SZ   32
#define C_IN   256
#define C_OUT  256
#define L_SZ   4096
#define KW     3
#define KDIM   (C_IN * KW)        // 768
#define NTILE  64                 // output positions per workgroup
#define LROWS  (NTILE + 2)        // 66 staged positions (halo for K=3, pad 1)
#define LDS_STRIDE 264            // halves; 528 bytes/row, 16B aligned

// ---------------------------------------------------------------------------
// Weight prep: scale = mean|w| per output channel (training-mode binary-conv
// scale); forward-numerics of STE weight == w, so w_eff = w * scale.
// Re-layout to fp16 [co][ tap*256 + ci ] so WMMA K-chunk kidx covers halves
// [32*kidx, 32*kidx+32) -- pure immediate-offset loads after unroll.
// ---------------------------------------------------------------------------
__global__ void prep_weights(const float* __restrict__ w,
                             _Float16* __restrict__ wh) {
  const int co = blockIdx.x;
  const int t  = threadIdx.x;
  const float* wr = w + co * KDIM;   // layout [co][ci*3 + kk]

  __shared__ float red[256];
  red[t] = fabsf(wr[t]) + fabsf(wr[t + 256]) + fabsf(wr[t + 512]);
  __syncthreads();
  for (int s = 128; s > 0; s >>= 1) {
    if (t < s) red[t] += red[t + s];
    __syncthreads();
  }
  const float scale = red[0] * (1.0f / (float)KDIM);

  _Float16* whr = wh + co * KDIM;
  for (int i = t; i < KDIM; i += 256) {
    const int ci = i / 3;
    const int kk = i - ci * 3;
    whr[kk * 256 + ci] = (_Float16)(wr[i] * scale);
  }
}

// ---------------------------------------------------------------------------
// Main kernel: one WG = one (batch b, 64-position tile). 8 waves.
// Wave w owns m-tiles {2w, 2w+1} (32 output channels) x all 4 n-tiles:
// zero intra-WG weight duplication, 2 A-frag + 4 B-frag loads per K-step.
// ---------------------------------------------------------------------------
__global__ void __launch_bounds__(256)
binary_block_main(const float* __restrict__ x,
                  const float* __restrict__ alpha,
                  const _Float16* __restrict__ wh,
                  const float* __restrict__ bias,
                  const float* __restrict__ beta,
                  const float* __restrict__ gamma,
                  const float* __restrict__ zeta,
                  float* __restrict__ out) {
  __shared__ _Float16 xT[LROWS * LDS_STRIDE];   // transposed: [pos][ci]

  const int tid = threadIdx.x;
  const int b   = blockIdx.x >> 6;          // 32 batches
  const int l0  = (blockIdx.x & 63) * NTILE;

  // ---- stage binarized (+/-1, zero-padded) transposed x tile into LDS ----
  const float* xb = x + (size_t)b * C_IN * L_SZ;
  for (int idx = tid; idx < C_IN * LROWS; idx += 256) {
    const int ci = idx / LROWS;
    const int j  = idx - ci * LROWS;        // staged row: global l = l0-1+j
    const int l  = l0 - 1 + j;
    _Float16 v = (_Float16)0.f;             // conv zero-padding
    if (l >= 0 && l < L_SZ) {
      const float xv = xb[ci * L_SZ + l];
      v = (xv >= alpha[ci]) ? (_Float16)1.f : (_Float16)(-1.f);
    }
    xT[j * LDS_STRIDE + ci] = v;
  }
  __syncthreads();

  // ---- WMMA compute ----
  const int wave  = tid >> 5;
  const int lane  = tid & 31;
  const int laneM = lane & 15;
  const int hi    = (lane >> 4) & 1;        // half-wave selector

  const int mt0 = wave * 2;                 // two 16-channel tiles per wave

  v8f acc[2][4] = {};                       // [m-tile][n-tile] f32 accum

  // A base pointers: row = output channel, K contiguous in memory.
  // lanes<16 take halves K0-7 / K16-23; lanes>=16 take K8-15 / K24-31.
  const _Float16* ap0 = wh + ((mt0 + 0) * 16 + laneM) * KDIM + hi * 8;
  const _Float16* ap1 = wh + ((mt0 + 1) * 16 + laneM) * KDIM + hi * 8;

  // B base pointers per n-tile: column N = laneM; lanes<16 hold K0-7 / K16-23,
  // lanes>=16 hold K8-15 / K24-31.
  const _Float16* bp[4];
#pragma unroll
  for (int n = 0; n < 4; ++n)
    bp[n] = &xT[(n * 16 + laneM) * LDS_STRIDE + hi * 8];

#pragma unroll
  for (int kidx = 0; kidx < 24; ++kidx) {
    const int koff = kidx * 32;             // halves into weight row
    const int kk   = kidx >> 3;             // conv tap 0..2
    const int ci0  = (kidx & 7) * 32;       // input-channel block
    const int boff = kk * LDS_STRIDE + ci0; // halves into staged tile

    const v8h a00 = *(const v8h*)(ap0 + koff);
    const v8h a01 = *(const v8h*)(ap0 + koff + 16);
    const v16h af0 = __builtin_shufflevector(a00, a01,
        0,1,2,3,4,5,6,7,8,9,10,11,12,13,14,15);
    const v8h a10 = *(const v8h*)(ap1 + koff);
    const v8h a11 = *(const v8h*)(ap1 + koff + 16);
    const v16h af1 = __builtin_shufflevector(a10, a11,
        0,1,2,3,4,5,6,7,8,9,10,11,12,13,14,15);

#pragma unroll
    for (int n = 0; n < 4; ++n) {
      // B row offset +kk: input position l + kk - 1 lands at staged row
      // (l - l0) + kk.
      const _Float16* bq = bp[n] + boff + kk * 0; // kk already in boff row
      const v8h b0 = *(const v8h*)(bq);
      const v8h b1 = *(const v8h*)(bq + 16);
      const v16h bf = __builtin_shufflevector(b0, b1,
          0,1,2,3,4,5,6,7,8,9,10,11,12,13,14,15);
      acc[0][n] = __builtin_amdgcn_wmma_f32_16x16x32_f16(
          false, af0, false, bf, (short)0, acc[0][n], false, false);
      acc[1][n] = __builtin_amdgcn_wmma_f32_16x16x32_f16(
          false, af1, false, bf, (short)0, acc[1][n], false, false);
    }
  }

  // ---- epilogue: bias + RPReLU, nontemporal store ----
  // C/D layout: VGPR r -> M = r + 8*hi, N = laneM.
#pragma unroll
  for (int mi = 0; mi < 2; ++mi) {
    const int co0 = (mt0 + mi) * 16 + hi * 8;   // 8 consecutive channels
    const float4 bq0 = *(const float4*)(bias  + co0);
    const float4 bq1 = *(const float4*)(bias  + co0 + 4);
    const float4 gq0 = *(const float4*)(gamma + co0);
    const float4 gq1 = *(const float4*)(gamma + co0 + 4);
    const float4 pq0 = *(const float4*)(beta  + co0);
    const float4 pq1 = *(const float4*)(beta  + co0 + 4);
    const float4 zq0 = *(const float4*)(zeta  + co0);
    const float4 zq1 = *(const float4*)(zeta  + co0 + 4);
    const float bs[8] = {bq0.x,bq0.y,bq0.z,bq0.w, bq1.x,bq1.y,bq1.z,bq1.w};
    const float gs[8] = {gq0.x,gq0.y,gq0.z,gq0.w, gq1.x,gq1.y,gq1.z,gq1.w};
    const float ps[8] = {pq0.x,pq0.y,pq0.z,pq0.w, pq1.x,pq1.y,pq1.z,pq1.w};
    const float zs[8] = {zq0.x,zq0.y,zq0.z,zq0.w, zq1.x,zq1.y,zq1.z,zq1.w};
#pragma unroll
    for (int n = 0; n < 4; ++n) {
      const int lcol = l0 + n * 16 + laneM;
#pragma unroll
      for (int r = 0; r < 8; ++r) {
        const float y  = acc[mi][n][r] + bs[r];
        const float yg = y - gs[r];
        const float o  = (y > gs[r]) ? (yg + zs[r]) : (ps[r] * yg + zs[r]);
        float* dst = out + (size_t)(b * C_OUT + co0 + r) * L_SZ + lcol;
        __builtin_nontemporal_store(o, dst);
      }
    }
  }
}

// ---------------------------------------------------------------------------
extern "C" void kernel_launch(void* const* d_in, const int* in_sizes, int n_in,
                              void* d_out, int out_size, void* d_ws, size_t ws_size,
                              hipStream_t stream) {
  const float* x     = (const float*)d_in[0];   // [32,256,4096]
  const float* alpha = (const float*)d_in[1];   // [256]
  const float* w     = (const float*)d_in[2];   // [256,256,3]
  const float* bias  = (const float*)d_in[3];   // [256]
  const float* beta  = (const float*)d_in[4];   // [256]
  const float* gamma = (const float*)d_in[5];   // [256]
  const float* zeta  = (const float*)d_in[6];   // [256]
  float* out = (float*)d_out;

  _Float16* wh = (_Float16*)d_ws;               // 256*768*2 = 384 KB scratch

  prep_weights<<<C_OUT, 256, 0, stream>>>(w, wh);
  binary_block_main<<<B_SZ * (L_SZ / NTILE), 256, 0, stream>>>(
      x, alpha, wh, bias, beta, gamma, zeta, out);
}
